// RoutingLayer_87402584473647
// MI455X (gfx1250) — compile-verified, matrix operand order checked by reading, not run on hardware
//
#include <hip/hip_runtime.h>
#include <hip/hip_bf16.h>

// ---------------------------------------------------------------------------
// RoutingLayer fused kernel for gfx1250 (MI455X).
// bf16 WMMA (16x16x32) with f32 accumulation; weights pre-converted to bf16.
// LDS: padded stride (260 dw/row) -> ds_load_b128 A-fragments, no xor math.
// ---------------------------------------------------------------------------

typedef __attribute__((ext_vector_type(16))) __bf16 v16bf;
typedef __attribute__((ext_vector_type(8)))  float  v8f;

union Frag {
    unsigned int u[8];
    v16bf        b;
};

#define D_DIM    512
#define S_DIM    2048
#define TILE_M   128            // rows per workgroup (8 waves x 16)
#define NWAVES   8              // 256 threads
#define LDS_STRIDE 260          // dwords per row (512 bf16 + 8 bf16 pad)
#define LDS_BYTES  (TILE_M * LDS_STRIDE * 4)   // 133,120 B (CDNA5: <=320KB/WG)

__device__ __forceinline__ unsigned int f2bf(float f) {
    unsigned int u = __float_as_uint(f);
    return (u + 0x7FFFu + ((u >> 16) & 1u)) >> 16;   // round-to-nearest-even
}
__device__ __forceinline__ float bf2f(unsigned int us) {
    return __uint_as_float(us << 16);
}

// --- prologue: convert f32 weights -> bf16 scratch (stays L2-resident) -----
__global__ void convert_bf16_kernel(const float* __restrict__ shareW,
                                    const float* __restrict__ langsW,
                                    const float* __restrict__ gateW1,
                                    unsigned short* __restrict__ shareB,
                                    unsigned short* __restrict__ langsB,
                                    unsigned short* __restrict__ gateB) {
    const long nA = (long)D_DIM * D_DIM;           // 256K
    const long nB = 8L * D_DIM * D_DIM;            // 2M
    const long total = nA + nB + nB;
    for (long i = (long)blockIdx.x * blockDim.x + threadIdx.x; i < total;
         i += (long)gridDim.x * blockDim.x) {
        if (i < nA)            shareB[i]          = (unsigned short)f2bf(shareW[i]);
        else if (i < nA + nB)  langsB[i - nA]     = (unsigned short)f2bf(langsW[i - nA]);
        else                   gateB[i - nA - nB] = (unsigned short)f2bf(gateW1[i - nA - nB]);
    }
}

// Fill LDS tile (TILE_M x 512 bf16, padded stride) from f32 rows [row0, +TILE_M)
__device__ __forceinline__ void fill_tile(unsigned int* lds,
                                          const float* __restrict__ src,
                                          int row0) {
    // float4 per iteration -> two packed-bf16 dwords (8-byte aligned b64 store)
    for (int i = threadIdx.x; i < TILE_M * 128; i += NWAVES * 32) {
        const int row = i >> 7;          // / (512/4)
        const int c4  = i & 127;
        const float4 v = *(const float4*)(src + (size_t)(row0 + row) * D_DIM + c4 * 4);
        unsigned int d0 = f2bf(v.x) | (f2bf(v.y) << 16);
        unsigned int d1 = f2bf(v.z) | (f2bf(v.w) << 16);
        unsigned int* p = lds + row * LDS_STRIDE + c4 * 2;
        p[0] = d0;
        p[1] = d1;
    }
}

// A fragment (16x32 bf16), row = lane%16, K split by half per ISA layout:
// elems 0..7 : K = k0 + half*8 + 0..7 ; elems 8..15 : K = k0 + 16 + half*8 + 0..7
// Padded stride keeps each 8-bf16 run contiguous & 16B aligned -> 2x ds_load_b128.
__device__ __forceinline__ v16bf load_afrag(const unsigned int* lds,
                                            int row, int k0, int half) {
    Frag f;
    const uint4* p = (const uint4*)(lds + row * LDS_STRIDE + half * 4 + (k0 >> 1));
    const uint4 q0 = p[0];               // K = k0 + half*8 + 0..7
    const uint4 q1 = p[2];               // K = k0 + 16 + half*8 + 0..7
    f.u[0] = q0.x; f.u[1] = q0.y; f.u[2] = q0.z; f.u[3] = q0.w;
    f.u[4] = q1.x; f.u[5] = q1.y; f.u[6] = q1.z; f.u[7] = q1.w;
    return f.b;
}

// B fragment (32x16): einsum "...d,ed->...e" means B[k][n] = W[n][k], so each
// lane (n = n_base + lane%16) reads contiguous K-runs of row n of W (bf16).
__device__ __forceinline__ v16bf load_bfrag(const unsigned short* __restrict__ W,
                                            int n, int k0, int half) {
    Frag f;
    const uint4 q0 = *(const uint4*)(W + (size_t)n * D_DIM + k0 + half * 8);
    const uint4 q1 = *(const uint4*)(W + (size_t)n * D_DIM + k0 + 16 + half * 8);
    f.u[0] = q0.x; f.u[1] = q0.y; f.u[2] = q0.z; f.u[3] = q0.w;
    f.u[4] = q1.x; f.u[5] = q1.y; f.u[6] = q1.z; f.u[7] = q1.w;
    return f.b;
}

__global__ __launch_bounds__(NWAVES * 32, 1)
void routing_fused_kernel(const float* __restrict__ old_x,
                          const float* __restrict__ x,
                          const int*   __restrict__ lang,
                          const unsigned short* __restrict__ shareB,
                          const unsigned short* __restrict__ langsB,
                          const unsigned short* __restrict__ gateB,
                          const float* __restrict__ gate_b1,
                          const float* __restrict__ gate_W2,
                          const float* __restrict__ gate_b2,
                          float* __restrict__ out) {
    extern __shared__ unsigned int lds[];           // 133 KB, reused A->B phase

    const int b    = blockIdx.y;
    const int row0 = blockIdx.x * TILE_M;
    const int lane = threadIdx.x & 31;
    const int wave = threadIdx.x >> 5;
    const int half = lane >> 4;                     // K-half / row-half select
    const int m    = lane & 15;                     // A row | B/C col within tile
    const int wrow = wave * 16;                     // wave's M-subtile base

    const int lg = lang[b];
    const unsigned short* Wl  = langsB + (size_t)lg * D_DIM * D_DIM;
    const unsigned short* W1  = gateB  + (size_t)lg * D_DIM * D_DIM;
    const float* b1  = gate_b1 + lg * D_DIM;
    const float* W2  = gate_W2 + lg * D_DIM;
    const float  b2v = gate_b2[lg];

    const float* oxb = old_x + (size_t)b * S_DIM * D_DIM;
    const float* xb  = x     + (size_t)b * S_DIM * D_DIM;

    const v8f zero8 = {0.f, 0.f, 0.f, 0.f, 0.f, 0.f, 0.f, 0.f};

    // ---------------- Phase A: gate logits over old_x ----------------------
    fill_tile(lds, oxb, row0);
    __syncthreads();

    float gsum[8];
#pragma unroll
    for (int j = 0; j < 8; ++j) gsum[j] = 0.f;

    for (int nc = 0; nc < 8; ++nc) {                // 8 chunks x 64 cols
        v8f acc[4];
#pragma unroll
        for (int u = 0; u < 4; ++u) acc[u] = zero8;

#pragma unroll 4
        for (int kt = 0; kt < 16; ++kt) {
            const int k0 = kt * 32;
            const v16bf a = load_afrag(lds, wrow + m, k0, half);
#pragma unroll
            for (int u = 0; u < 4; ++u) {
                const int n = (nc * 4 + u) * 16 + m;
                const v16bf bw = load_bfrag(W1, n, k0, half);
                acc[u] = __builtin_amdgcn_wmma_f32_16x16x32_bf16(
                    false, a, false, bw, (short)0, acc[u], false, false);
            }
        }
        // fuse relu(+b1) + old_x, then partial dot with W2
#pragma unroll
        for (int u = 0; u < 4; ++u) {
            const int e   = (nc * 4 + u) * 16 + m;
            const float b1e = b1[e];
            const float w2e = W2[e];
#pragma unroll
            for (int j = 0; j < 8; ++j) {
                const int r = wrow + half * 8 + j;  // C layout: row=half*8+j
                const unsigned int dw = lds[r * LDS_STRIDE + (e >> 1)];
                const float ox = bf2f((e & 1) ? (dw >> 16) : (dw & 0xffffu));
                float h = acc[u][j] + b1e;
                h = h > 0.f ? h : 0.f;
                gsum[j] += (h + ox) * w2e;
            }
        }
    }

    // butterfly-reduce each row's dot over the 16 lanes of this half-wave
    float p[8];
#pragma unroll
    for (int j = 0; j < 8; ++j) {
        float s = gsum[j];
        s += __shfl_xor(s, 1, 16);
        s += __shfl_xor(s, 2, 16);
        s += __shfl_xor(s, 4, 16);
        s += __shfl_xor(s, 8, 16);
        p[j] = 1.f / (1.f + __expf(-(s + b2v)));    // sigmoid(gate_logit)
    }

    // ---------------- Phase B: shared + expert GEMMs, blended write --------
    __syncthreads();
    fill_tile(lds, xb, row0);
    __syncthreads();

    for (int nc = 0; nc < 8; ++nc) {
        v8f accS[4], accL[4];
#pragma unroll
        for (int u = 0; u < 4; ++u) { accS[u] = zero8; accL[u] = zero8; }

#pragma unroll 4
        for (int kt = 0; kt < 16; ++kt) {
            const int k0 = kt * 32;
            const v16bf a = load_afrag(lds, wrow + m, k0, half);
#pragma unroll
            for (int u = 0; u < 4; ++u) {
                const int n = (nc * 4 + u) * 16 + m;
                const v16bf bs = load_bfrag(shareB, n, k0, half);
                accS[u] = __builtin_amdgcn_wmma_f32_16x16x32_bf16(
                    false, a, false, bs, (short)0, accS[u], false, false);
                const v16bf bl = load_bfrag(Wl, n, k0, half);
                accL[u] = __builtin_amdgcn_wmma_f32_16x16x32_bf16(
                    false, a, false, bl, (short)0, accL[u], false, false);
            }
        }
#pragma unroll
        for (int u = 0; u < 4; ++u) {
            const int e = (nc * 4 + u) * 16 + m;
#pragma unroll
            for (int j = 0; j < 8; ++j) {
                const int r = row0 + wrow + half * 8 + j;
                const float pv  = p[j];
                const float val = accS[u][j] * (1.f - pv) + accL[u][j] * pv;
                out[((size_t)b * S_DIM + r) * D_DIM + e] = val;
            }
        }
    }
}

// ---------------------------------------------------------------------------
extern "C" void kernel_launch(void* const* d_in, const int* in_sizes, int n_in,
                              void* d_out, int out_size, void* d_ws, size_t ws_size,
                              hipStream_t stream) {
    const float* old_x   = (const float*)d_in[0];
    const float* x       = (const float*)d_in[1];
    const int*   lang    = (const int*)  d_in[2];
    const float* share_W = (const float*)d_in[3];
    const float* langs_W = (const float*)d_in[4];
    const float* gate_W1 = (const float*)d_in[5];
    const float* gate_b1 = (const float*)d_in[6];
    const float* gate_W2 = (const float*)d_in[7];
    const float* gate_b2 = (const float*)d_in[8];
    float* out = (float*)d_out;

    // bf16 weight scratch in d_ws: share (512KB) | langs (4MB) | gateW1 (4MB)
    unsigned short* shareB = (unsigned short*)d_ws;
    unsigned short* langsB = shareB + (size_t)D_DIM * D_DIM;
    unsigned short* gateB  = langsB + (size_t)8 * D_DIM * D_DIM;

    convert_bf16_kernel<<<dim3(2048), dim3(256), 0, stream>>>(
        share_W, langs_W, gate_W1, shareB, langsB, gateB);

    dim3 grid(S_DIM / TILE_M, 32);                  // 16 S-tiles x 32 batches
    routing_fused_kernel<<<grid, dim3(NWAVES * 32), LDS_BYTES, stream>>>(
        old_x, x, lang, shareB, langsB, gateB, gate_b1, gate_W2, gate_b2, out);
}
